// MultiHeadAttentionBlock_21474836480379
// MI455X (gfx1250) — matrix operand order repaired and checked by reading, not compiled, not run on hardware
//
#include <hip/hip_runtime.h>

// ---------------------------------------------------------------------------
// MultiHeadAttentionBlock (diagonal-softmax variant) for MI455X / gfx1250
// B=4, S=2048, D=1024, H=16, DQ=64
//  pass 1: pack q/k/v -> bf16, weights -> bf16 B^T
//  pass 2: Q/K/V projections: bf16 WMMA GEMM, B-tile double-buffered in LDS
//          via global_load_async_to_lds_b128 (ASYNCcnt)
//  pass 3: masked-softmax row stats via WMMA score tiles with K-tile
//          double-buffered in LDS; only the softmax DIAGONAL scales V.
// ---------------------------------------------------------------------------

typedef __bf16 bf16_t;
typedef bf16_t v16bf __attribute__((ext_vector_type(16)));
typedef bf16_t v8bf  __attribute__((ext_vector_type(8)));
typedef bf16_t v4bf  __attribute__((ext_vector_type(4)));
typedef float  v8f   __attribute__((ext_vector_type(8)));

#define BB 4
#define SS 2048
#define DD 1024
#define HH 16
#define DQV 64
static constexpr float kScale  = 0.125f;   // 1/sqrt(64)
static constexpr float kNegBig = -1e9f;

__device__ __forceinline__ bf16_t f2bf(float f) {
  unsigned u = __float_as_uint(f);
  unsigned r = (u + 0x7FFFu + ((u >> 16) & 1u)) >> 16;   // RNE
  unsigned short s = (unsigned short)r;
  return __builtin_bit_cast(bf16_t, s);
}
__device__ __forceinline__ float bf2f(bf16_t b) {
  unsigned short s = __builtin_bit_cast(unsigned short, b);
  return __uint_as_float(((unsigned)s) << 16);
}

// --- CDNA5 async global->LDS copies (ASYNCcnt) -----------------------------
__device__ __forceinline__ void async_load_b128(unsigned lds_off, const void* gaddr) {
  asm volatile("global_load_async_to_lds_b128 %0, %1, off"
               :: "v"(lds_off), "v"(gaddr) : "memory");
}
__device__ __forceinline__ void async_load_b64(unsigned lds_off, const void* gaddr) {
  asm volatile("global_load_async_to_lds_b64 %0, %1, off"
               :: "v"(lds_off), "v"(gaddr) : "memory");
}
__device__ __forceinline__ void wait_async0() {
  asm volatile("s_wait_asynccnt 0" ::: "memory");
}

// ---------------------------------------------------------------------------
// Pack an fp32 tensor to bf16 (flat), 4 elements per thread.
// ---------------------------------------------------------------------------
__global__ void pack_input_kernel(const float* __restrict__ x,
                                  bf16_t* __restrict__ y) {
  size_t i = ((size_t)blockIdx.x * blockDim.x + threadIdx.x) * 4;
  float4 f = *(const float4*)(x + i);
  v4bf o;
  o[0] = f2bf(f.x); o[1] = f2bf(f.y); o[2] = f2bf(f.z); o[3] = f2bf(f.w);
  *(v4bf*)(y + i) = o;
}

// Weights (H, D, DQ) fp32 -> Bt[h*DQ+q][d] bf16 (B^T: contiguous B-fragments)
__global__ void pack_weight_kernel(const float* __restrict__ w,
                                   bf16_t* __restrict__ wt) {
  int idx = blockIdx.x * blockDim.x + threadIdx.x;
  int d  = idx & (DD - 1);
  int n  = idx >> 10;            // h*DQ + q
  int h  = n >> 6;
  int qi = n & 63;
  wt[(size_t)n * DD + d] = f2bf(w[((size_t)h * DD + d) * DQV + qi]);
}

// ---------------------------------------------------------------------------
// Projection GEMM: C[8192][1024] = A[8192][1024]bf16 x B[1024][1024]bf16.
// Block: 8 waves = 128 rows x 64 cols. The 64-col B strip is shared by all
// waves and double-buffered in LDS via async B128 copies (4KB per k-step).
// ---------------------------------------------------------------------------
template <int STORE_F32>
__global__ void proj_gemm_kernel(const bf16_t* __restrict__ A,
                                 const bf16_t* __restrict__ Bt,
                                 bf16_t* __restrict__ outB,
                                 float* __restrict__ outF) {
  __shared__ bf16_t Bs[2][64 * 32];        // [buf][col][k] 4KB each

  const int tid  = threadIdx.x;
  const int lane = tid & 31;
  const int l16  = lane & 15;
  const int half = lane >> 4;
  const int wave = tid >> 5;

  const int row0 = (blockIdx.x >> 4) * 128 + wave * 16;
  const int n0   = (blockIdx.x & 15) * 64;

  // async stage: 256 threads x 16B cover the 64x32 bf16 tile
  const int c   = tid >> 2;                 // 0..63 column of strip
  const int seg = tid & 3;                  // 16B segment of the 64B row
  const bf16_t* gB = Bt + (size_t)(n0 + c) * DD + seg * 8;
  const unsigned ldsB = (unsigned)(c * 64 + seg * 16);  // bytes in one buffer

  const bf16_t* arow = A + (size_t)(row0 + l16) * DD + (half << 4);

  v8f acc0 = {}, acc1 = {}, acc2 = {}, acc3 = {};

  async_load_b128(ldsB, gB);                // k-tile 0 -> buf 0

  for (int k0 = 0, it = 0; k0 < DD; k0 += 32, ++it) {
    const int cur = it & 1;
    wait_async0();
    __syncthreads();
    if (k0 + 32 < DD)
      async_load_b128((unsigned)((cur ^ 1) * 4096) + ldsB, gB + k0 + 32);

    // A fragment: lanes 0-15 K=k0+0..15, lanes 16-31 K=k0+16..31
    v16bf a;
    {
      v8bf lo = *(const v8bf*)(arow + k0);
      v8bf hi = *(const v8bf*)(arow + k0 + 8);
#pragma unroll
      for (int i = 0; i < 8; ++i) { a[i] = lo[i]; a[i + 8] = hi[i]; }
    }
    const bf16_t* bs = &Bs[cur][0];
    v16bf b0 = *(const v16bf*)(bs + (0 * 16 + l16) * 32 + half * 16);
    v16bf b1 = *(const v16bf*)(bs + (1 * 16 + l16) * 32 + half * 16);
    v16bf b2 = *(const v16bf*)(bs + (2 * 16 + l16) * 32 + half * 16);
    v16bf b3 = *(const v16bf*)(bs + (3 * 16 + l16) * 32 + half * 16);

    acc0 = __builtin_amdgcn_wmma_f32_16x16x32_bf16(false, a, false, b0, (short)0, acc0, false, false);
    acc1 = __builtin_amdgcn_wmma_f32_16x16x32_bf16(false, a, false, b1, (short)0, acc1, false, false);
    acc2 = __builtin_amdgcn_wmma_f32_16x16x32_bf16(false, a, false, b2, (short)0, acc2, false, false);
    acc3 = __builtin_amdgcn_wmma_f32_16x16x32_bf16(false, a, false, b3, (short)0, acc3, false, false);
  }

  // Scatter C fragment into [b,h,s,dq]
#pragma unroll
  for (int r = 0; r < 8; ++r) {
    const int Mrow = row0 + (half ? r + 8 : r);
    const int bb = Mrow >> 11;
    const int s  = Mrow & (SS - 1);
    float vals[4] = {acc0[r], acc1[r], acc2[r], acc3[r]};
#pragma unroll
    for (int j = 0; j < 4; ++j) {
      const int ncol = n0 + j * 16 + l16;
      const int h  = ncol >> 6;
      const int qi = ncol & 63;
      const size_t oidx = (((size_t)bb * HH + h) * SS + s) * DQV + qi;
      if (STORE_F32) outF[oidx] = vals[j];
      else           outB[oidx] = f2bf(vals[j]);
    }
  }
}

// ---------------------------------------------------------------------------
// Stage 3: block = 8 waves = 128 query rows of the SAME (b,h). K tiles
// (16x64 bf16, 2KB) shared by all waves, double-buffered via async B64.
// Per 16x16 score tile: 2 WMMAs, masked online softmax (row max + sum).
// Diagonal numerator recomputed as a 64-MAC dot; out = w * V.
// ---------------------------------------------------------------------------
__global__ void attn_diag_kernel(const bf16_t* __restrict__ Qp,
                                 const bf16_t* __restrict__ Kp,
                                 const float* __restrict__ Vp,
                                 const int* __restrict__ mask,
                                 float* __restrict__ out) {
  __shared__ bf16_t Ks[2][16 * 64];        // [buf][t][dq] 2KB each

  const int tid  = threadIdx.x;
  const int lane = tid & 31;
  const int l16  = lane & 15;
  const int half = lane >> 4;
  const int wave = tid >> 5;

  const int wid = blockIdx.x * 8 + wave;   // 0..8191
  const int b  = wid >> 11;
  const int h  = (wid >> 7) & 15;
  const int sb = wid & 127;
  const int row0 = sb << 4;

  const size_t base = (((size_t)b * HH) + h) * (size_t)SS * DQV;
  const int* mbase = mask + (size_t)b * SS * SS;

  // async stage: 256 threads x 8B cover the 16x64 bf16 K tile
  const int tr  = tid >> 4;                // 0..15 row of tile
  const int seg = tid & 15;                // 8B segment of 128B row
  const bf16_t* gK = Kp + base + (size_t)tr * DQV + seg * 4;
  const unsigned ldsK = (unsigned)(tr * 128 + seg * 8);

  // A fragments: 16 Q rows, K-dim 0..31 (a0) and 32..63 (a1)
  const bf16_t* qrow = Qp + base + (size_t)(row0 + l16) * DQV;
  const int off = half << 4;
  v16bf a0, a1;
  {
    v8bf lo = *(const v8bf*)(qrow + off);
    v8bf hi = *(const v8bf*)(qrow + off + 8);
#pragma unroll
    for (int i = 0; i < 8; ++i) { a0[i] = lo[i]; a0[i + 8] = hi[i]; }
  }
  {
    v8bf lo = *(const v8bf*)(qrow + 32 + off);
    v8bf hi = *(const v8bf*)(qrow + 32 + off + 8);
#pragma unroll
    for (int i = 0; i < 8; ++i) { a1[i] = lo[i]; a1[i + 8] = hi[i]; }
  }

  float m[8], p[8];
#pragma unroll
  for (int r = 0; r < 8; ++r) { m[r] = -3.0e38f; p[r] = 0.0f; }

  async_load_b64(ldsK, gK);                // t-tile 0 -> buf 0

  for (int t0 = 0, it = 0; t0 < SS; t0 += 16, ++it) {
    const int cur = it & 1;
    wait_async0();
    __syncthreads();
    if (t0 + 16 < SS)
      async_load_b64((unsigned)((cur ^ 1) * 2048) + ldsK,
                     gK + (size_t)(t0 + 16) * DQV);

    const bf16_t* ks = &Ks[cur][0];
    v16bf b0 = *(const v16bf*)(ks + l16 * 64 + half * 16);
    v16bf b1 = *(const v16bf*)(ks + l16 * 64 + half * 16 + 32);

    v8f acc = {};
    acc = __builtin_amdgcn_wmma_f32_16x16x32_bf16(false, a0, false, b0, (short)0, acc, false, false);
    acc = __builtin_amdgcn_wmma_f32_16x16x32_bf16(false, a1, false, b1, (short)0, acc, false, false);

#pragma unroll
    for (int r = 0; r < 8; ++r) {
      const int row = row0 + (half ? r + 8 : r);
      const int mv = mbase[(size_t)row * SS + t0 + l16];
      const float val = mv ? acc[r] * kScale : kNegBig;
      const float nm = fmaxf(m[r], val);
      p[r] = p[r] * __expf(m[r] - nm) + __expf(val - nm);
      m[r] = nm;
    }
  }

  // Combine the 16 lanes sharing each C-fragment row (xor stays in-half).
#pragma unroll
  for (int r = 0; r < 8; ++r) {
#pragma unroll
    for (int sft = 1; sft < 16; sft <<= 1) {
      const float om = __shfl_xor(m[r], sft, 32);
      const float op = __shfl_xor(p[r], sft, 32);
      const float nm = fmaxf(m[r], om);
      p[r] = p[r] * __expf(m[r] - nm) + op * __expf(om - nm);
      m[r] = nm;
    }
  }

  // Diagonal weight + output: each lane writes 4 of the 64 dq values per row.
#pragma unroll
  for (int r = 0; r < 8; ++r) {
    const int row = row0 + (half ? r + 8 : r);
    const bf16_t* qr  = Qp + base + (size_t)row * DQV;
    const bf16_t* krr = Kp + base + (size_t)row * DQV;
    float d = 0.0f;
#pragma unroll
    for (int c = 0; c < 8; ++c) {
      v8bf qa = *(const v8bf*)(qr  + c * 8);
      v8bf ka = *(const v8bf*)(krr + c * 8);
#pragma unroll
      for (int i = 0; i < 8; ++i) d += bf2f(qa[i]) * bf2f(ka[i]);
    }
    const int mv = mbase[(size_t)row * SS + row];
    const float val = mv ? d * kScale : kNegBig;
    const float w = __expf(val - m[r]) / p[r];

    const size_t o = base + (size_t)row * DQV + (size_t)l16 * 4;
    float4 vv = *(const float4*)(Vp + o);
    float4 ov;
    ov.x = w * vv.x; ov.y = w * vv.y; ov.z = w * vv.z; ov.w = w * vv.w;
    *(float4*)(out + o) = ov;
  }
}

// ---------------------------------------------------------------------------
extern "C" void kernel_launch(void* const* d_in, const int* in_sizes, int n_in,
                              void* d_out, int out_size, void* d_ws, size_t ws_size,
                              hipStream_t stream) {
  const float* q  = (const float*)d_in[0];
  const float* k  = (const float*)d_in[1];
  const float* v  = (const float*)d_in[2];
  const int*   mk = (const int*)d_in[3];
  const float* wq = (const float*)d_in[4];
  const float* wk = (const float*)d_in[5];
  const float* wv = (const float*)d_in[6];
  float* out = (float*)d_out;

  // Workspace layout (~118 MB)
  char* p = (char*)d_ws;
  auto take = [&](size_t bytes) {
    char* r = p;
    p += (bytes + 255) & ~(size_t)255;
    return r;
  };
  const size_t wt_bytes = (size_t)HH * DQV * DD * sizeof(bf16_t);       // 2 MB
  const size_t in_bytes = (size_t)BB * SS * DD * sizeof(bf16_t);        // 16 MB
  const size_t qk_bytes = (size_t)BB * HH * SS * DQV * sizeof(bf16_t);  // 16 MB
  const size_t v_bytes  = (size_t)BB * HH * SS * DQV * sizeof(float);   // 32 MB

  bf16_t* wqT = (bf16_t*)take(wt_bytes);
  bf16_t* wkT = (bf16_t*)take(wt_bytes);
  bf16_t* wvT = (bf16_t*)take(wt_bytes);
  bf16_t* qb  = (bf16_t*)take(in_bytes);
  bf16_t* kb  = (bf16_t*)take(in_bytes);
  bf16_t* vb  = (bf16_t*)take(in_bytes);
  bf16_t* Qp  = (bf16_t*)take(qk_bytes);
  bf16_t* Kp  = (bf16_t*)take(qk_bytes);
  float*  Vp  = (float*)take(v_bytes);

  const dim3 blk(256);

  // 1) Pack inputs + weights to bf16
  const int npack = (BB * SS * DD) / 4 / 256;      // 8192 blocks
  pack_input_kernel<<<npack, blk, 0, stream>>>(q, qb);
  pack_input_kernel<<<npack, blk, 0, stream>>>(k, kb);
  pack_input_kernel<<<npack, blk, 0, stream>>>(v, vb);
  pack_weight_kernel<<<(HH * DQV * DD) / 256, blk, 0, stream>>>(wq, wqT);
  pack_weight_kernel<<<(HH * DQV * DD) / 256, blk, 0, stream>>>(wk, wkT);
  pack_weight_kernel<<<(HH * DQV * DD) / 256, blk, 0, stream>>>(wv, wvT);

  // 2) Projections: 64 M-blocks x 16 N-strips = 1024 blocks
  proj_gemm_kernel<0><<<1024, blk, 0, stream>>>(qb, wqT, Qp, nullptr);
  proj_gemm_kernel<0><<<1024, blk, 0, stream>>>(kb, wkT, Kp, nullptr);
  proj_gemm_kernel<1><<<1024, blk, 0, stream>>>(vb, wvT, nullptr, Vp);

  // 3) Diagonal-softmax attention: 8192 waves / 8 per block
  attn_diag_kernel<<<1024, blk, 0, stream>>>(Qp, Kp, Vp, mk, out);
}
